// AttentionLayers_29755533427252
// MI455X (gfx1250) — matrix operand order repaired and checked by reading, not compiled, not run on hardware
//
#include <hip/hip_runtime.h>
#include <hip/hip_bf16.h>
#include <math.h>
#include <float.h>

// ---------------- problem constants ----------------
#define BB     4
#define NN     1024
#define DIMD   1024
#define HH     16
#define DHD    64
#define INNER  1024          // H*DH
#define MEMK   4
#define JJ     (NN + MEMK)   // 1028 valid keys
#define JP     1056          // padded to multiple of 32
#define DFF    4096
#define TOK    (BB * NN)     // 4096 token rows
#define DEPTHL 2
#define TOPKK  8
#define SCALEF 0.125f        // DH^-0.5

typedef __bf16 bf16_t;
typedef __attribute__((ext_vector_type(16))) __bf16 v16bf;
typedef __attribute__((ext_vector_type(8)))  float  v8f;

// low 32 bits of a generic pointer into LDS == wave-logical LDS byte offset
// (ISA 10.2: LDS aperture maps by truncating to addr[31:0])
#define LDS_ADDR32(p) ((unsigned)(unsigned long long)(p))

// ---------------- elementwise f32 -> bf16 cast ----------------
__global__ void cast_bf16_kernel(const float* __restrict__ in,
                                 bf16_t* __restrict__ out, long long n) {
  long long i = (long long)blockIdx.x * blockDim.x + threadIdx.x;
  if (i < n) out[i] = (bf16_t)in[i];
}

// ---------------- layernorm over DIM=1024, bf16 out ----------------
__global__ __launch_bounds__(256)
void ln_kernel(const float* __restrict__ x, const float* __restrict__ g,
               const float* __restrict__ b, bf16_t* __restrict__ out) {
  __shared__ float red[256];
  int row = blockIdx.x;
  const float* xr = x + (long long)row * DIMD;
  float v[4];
  float s = 0.f;
#pragma unroll
  for (int i = 0; i < 4; ++i) { v[i] = xr[threadIdx.x + i * 256]; s += v[i]; }
  red[threadIdx.x] = s; __syncthreads();
  for (int o = 128; o > 0; o >>= 1) {
    if (threadIdx.x < o) red[threadIdx.x] += red[threadIdx.x + o];
    __syncthreads();
  }
  float mu = red[0] * (1.0f / DIMD);
  __syncthreads();
  s = 0.f;
#pragma unroll
  for (int i = 0; i < 4; ++i) { float d = v[i] - mu; s += d * d; }
  red[threadIdx.x] = s; __syncthreads();
  for (int o = 128; o > 0; o >>= 1) {
    if (threadIdx.x < o) red[threadIdx.x] += red[threadIdx.x + o];
    __syncthreads();
  }
  float rstd = rsqrtf(red[0] * (1.0f / DIMD) + 1e-5f);
  bf16_t* orow = out + (long long)row * DIMD;
#pragma unroll
  for (int i = 0; i < 4; ++i) {
    int c = threadIdx.x + i * 256;
    orow[c] = (bf16_t)((v[i] - mu) * rstd * g[c] + b[c]);
  }
}

// ---------------- generic bf16 WMMA GEMM ----------------
// C = act(alpha * A @ B + bias) + resid ; A[M,K] row-major bf16.
// opB==0: B[K,N] row-major (VALU transpose-scatter into LDS);
// opB==1: B stored [N,K] row-major (B^T) -> async global->LDS copy.
// A tile always staged with async global->LDS copies (ASYNCcnt).
// OOB rows/cols skip their loads: stale LDS only feeds discarded outputs.
// LDS layouts (fragment-major, 40-elem padded rows):
//   As[r][k]  : A fragment = contiguous 8-elem runs at k = 8*half (+16)
//   Bs[n][k]  : B fragment = contiguous 8-elem runs at k = 8*half (+16)
#define BM 128
#define BN 128
#define BK 32
#define LDP 40   // padded LDS row stride in elements (80B, 16B-aligned)

__global__ __launch_bounds__(256)
void gemm_wmma(const bf16_t* __restrict__ A, long long sAz, int lda,
               const bf16_t* __restrict__ Bm, long long sBz, int ldb, int opB,
               float* __restrict__ Cf, bf16_t* __restrict__ Cb,
               int zdivH, long long sCzb, long long sCzh, int ldc,
               const float* __restrict__ bias, const float* __restrict__ resid,
               int M, int N, int K, float alpha, int act) {
  __shared__ alignas(16) bf16_t As[BM * LDP];
  __shared__ alignas(16) bf16_t Bs[BN * LDP];

  int z = blockIdx.z;
  const bf16_t* Az = A + (long long)z * sAz;
  const bf16_t* Bz = Bm + (long long)z * sBz;
  long long cOff = (long long)(z / zdivH) * sCzb + (long long)(z % zdivH) * sCzh;

  int tid  = threadIdx.x;
  int lane = tid & 31;
  int wave = tid >> 5;
  int half = lane >> 4;
  int lr   = lane & 15;
  int waveM = wave >> 1;   // 0..3 : 32 rows each
  int waveN = wave & 1;    // 0..1 : 64 cols each
  int bm0 = blockIdx.x * BM;
  int bn0 = blockIdx.y * BN;

  v8f acc[2][4];
#pragma unroll
  for (int mi = 0; mi < 2; ++mi)
#pragma unroll
    for (int ni = 0; ni < 4; ++ni)
#pragma unroll
      for (int e = 0; e < 8; ++e) acc[mi][ni][e] = 0.f;

  for (int k0 = 0; k0 < K; k0 += BK) {
    // ---- stage A tile: async global->LDS, 16B per lane ----
#pragma unroll
    for (int it = 0; it < 2; ++it) {
      int e = (tid + it * 256) * 8;
      int r = e >> 5, c = e & 31;
      int gr = bm0 + r;
      if (gr < M) {
        unsigned lds = LDS_ADDR32(&As[r * LDP + c]);
        unsigned long long ga =
            (unsigned long long)(Az + (long long)gr * lda + (k0 + c));
        asm volatile("global_load_async_to_lds_b128 %0, %1, off"
                     :: "v"(lds), "v"(ga) : "memory");
      }
    }
    // ---- stage B tile into fragment-major Bs[n][k] ----
    if (opB == 1) {
      // B^T rows are K-contiguous: async global->LDS copy
#pragma unroll
      for (int it = 0; it < 2; ++it) {
        int e = (tid + it * 256) * 8;
        int n = e >> 5, k = e & 31;
        int gn = bn0 + n;
        if (gn < N) {
          unsigned lds = LDS_ADDR32(&Bs[n * LDP + k]);
          unsigned long long ga =
              (unsigned long long)(Bz + (long long)gn * ldb + (k0 + k));
          asm volatile("global_load_async_to_lds_b128 %0, %1, off"
                       :: "v"(lds), "v"(ga) : "memory");
        }
      }
    } else {
      // global row k, 8 consecutive n (coalesced) -> transpose-scatter to LDS
#pragma unroll
      for (int it = 0; it < 2; ++it) {
        int e = (tid + it * 256) * 8;
        int k = e >> 7, n8 = e & 127;
        int gn0 = bn0 + n8;
        const bf16_t* src = Bz + (long long)(k0 + k) * ldb + gn0;
        if (gn0 + 7 < N) {
          int4 val = *(const int4*)src;
          const bf16_t* t = (const bf16_t*)&val;
#pragma unroll
          for (int q = 0; q < 8; ++q) Bs[(n8 + q) * LDP + k] = t[q];
        } else {
#pragma unroll
          for (int q = 0; q < 8; ++q)
            if (gn0 + q < N) Bs[(n8 + q) * LDP + k] = src[q];
        }
      }
    }
    // prefetch next K-tile into cache while this tile computes (global_prefetch_b8)
    if (k0 + BK < K) {
      int e = tid * 8;
      int r = e >> 5, c = e & 31;
      int gr = bm0 + r;
      if (gr < M)
        __builtin_prefetch(Az + (long long)gr * lda + (k0 + BK + c), 0, 3);
      if (opB == 0) {
        int k = e >> 7, n8 = e & 127;
        if (bn0 + n8 < N)
          __builtin_prefetch(Bz + (long long)(k0 + BK + k) * ldb + bn0 + n8, 0, 3);
      } else {
        int n = e >> 5, k = e & 31;
        if (bn0 + n < N)
          __builtin_prefetch(Bz + (long long)(bn0 + n) * ldb + (k0 + BK + k), 0, 3);
      }
    }
    // wait for this wave's async LDS writes, then workgroup barrier
    asm volatile("s_wait_asynccnt 0x0" ::: "memory");
    __syncthreads();

    // ---- load fragments as b128 pairs and issue WMMA ----
    v16bf afr[2], bfr[4];
#pragma unroll
    for (int mi = 0; mi < 2; ++mi) {
      int off = (waveM * 32 + mi * 16 + lr) * LDP + half * 8;
      ((int4*)&afr[mi])[0] = *(const int4*)&As[off];
      ((int4*)&afr[mi])[1] = *(const int4*)&As[off + 16];
    }
#pragma unroll
    for (int ni = 0; ni < 4; ++ni) {
      int off = (waveN * 64 + ni * 16 + lr) * LDP + half * 8;
      ((int4*)&bfr[ni])[0] = *(const int4*)&Bs[off];
      ((int4*)&bfr[ni])[1] = *(const int4*)&Bs[off + 16];
    }
#pragma unroll
    for (int mi = 0; mi < 2; ++mi)
#pragma unroll
      for (int ni = 0; ni < 4; ++ni)
        acc[mi][ni] = __builtin_amdgcn_wmma_f32_16x16x32_bf16(
            false, afr[mi], false, bfr[ni], (short)0, acc[mi][ni], false, false);
    __syncthreads();
  }

  // ---- epilogue: scale, bias, GELU, residual, store f32 and/or bf16 ----
#pragma unroll
  for (int mi = 0; mi < 2; ++mi) {
#pragma unroll
    for (int ni = 0; ni < 4; ++ni) {
      int gn = bn0 + waveN * 64 + ni * 16 + lr;
      if (gn >= N) continue;
#pragma unroll
      for (int v = 0; v < 8; ++v) {
        int gm = bm0 + waveM * 32 + mi * 16 + (v + 8 * half);
        if (gm >= M) continue;
        float val = acc[mi][ni][v] * alpha;
        if (bias) val += bias[gn];
        if (act == 1) val = 0.5f * val * (1.0f + erff(val * 0.70710678118f));
        long long ci = cOff + (long long)gm * ldc + gn;
        if (resid) val += resid[ci];
        if (Cf) Cf[ci] = val;
        if (Cb) Cb[ci] = (bf16_t)val;
      }
    }
  }
}

// ---------------- q reshape: [b,i,h*64+d] -> [b,h,i,d] ----------------
__global__ void split_q_kernel(const bf16_t* __restrict__ qin,
                               bf16_t* __restrict__ qh) {
  long long idx = (long long)blockIdx.x * 256 + threadIdx.x;
  long long total = (long long)BB * HH * NN * DHD;
  if (idx >= total) return;
  int d = (int)(idx & (DHD - 1));
  long long t = idx >> 6;
  int i = (int)(t % NN); t /= NN;
  int h = (int)(t % HH);
  int b = (int)(t / HH);
  qh[idx] = qin[((long long)(b * NN + i)) * INNER + h * DHD + d];
}

// ---------------- build K/V with learned memory prefix + zero pad ----------------
__global__ void build_kv_kernel(const bf16_t* __restrict__ kin,
                                const bf16_t* __restrict__ vin,
                                const float* __restrict__ mem_k,
                                const float* __restrict__ mem_v, int layer,
                                bf16_t* __restrict__ Kc, bf16_t* __restrict__ Vc) {
  long long idx = (long long)blockIdx.x * 256 + threadIdx.x;
  long long total = (long long)BB * HH * JP * DHD;
  if (idx >= total) return;
  int d = (int)(idx & (DHD - 1));
  long long t = idx >> 6;
  int j = (int)(t % JP); t /= JP;
  int h = (int)(t % HH);
  int b = (int)(t / HH);
  bf16_t kv, vv;
  if (j < MEMK) {
    long long mo = (((long long)layer * HH + h) * MEMK + j) * DHD + d;
    kv = (bf16_t)mem_k[mo];
    vv = (bf16_t)mem_v[mo];
  } else if (j < JJ) {
    long long so = ((long long)(b * NN + (j - MEMK))) * INNER + h * DHD + d;
    kv = kin[so];
    vv = vin[so];
  } else {
    kv = (bf16_t)0.f;
    vv = (bf16_t)0.f;
  }
  Kc[idx] = kv;
  Vc[idx] = vv;
}

// ---------------- talking heads: s'[k] = sum_h s[h] * p[h,k], in place ----------------
template <typename T>
__global__ __launch_bounds__(256)
void talking_kernel(T* __restrict__ s, const float* __restrict__ p) {
  __shared__ float ps[HH * HH];
  ps[threadIdx.x] = p[threadIdx.x];   // blockDim.x == 256 == H*H
  __syncthreads();
  long long idx = (long long)blockIdx.x * 256 + threadIdx.x;
  long long total = (long long)BB * NN * JP;
  if (idx >= total) return;
  int j = (int)(idx % JP);
  long long t = idx / JP;
  int i = (int)(t % NN);
  int b = (int)(t / NN);
  long long base = ((long long)b * HH * NN + i) * JP + j;
  long long str = (long long)NN * JP;
  float v[HH];
#pragma unroll
  for (int h = 0; h < HH; ++h) v[h] = (float)s[base + h * str];
#pragma unroll
  for (int k = 0; k < HH; ++k) {
    float o = 0.f;
#pragma unroll
    for (int h = 0; h < HH; ++h) o += v[h] * ps[h * HH + k];
    s[base + k * str] = (T)o;
  }
}

// ---------------- causal mask + top-8 threshold + softmax, bf16 probs out ----------------
__global__ __launch_bounds__(256)
void topk_softmax_kernel(const float* __restrict__ dots,
                         bf16_t* __restrict__ probs) {
  __shared__ float sv[256];
  __shared__ int   si[256];
  int row = blockIdx.x;          // over B*H*N rows
  int i = row % NN;              // query index
  const float* dr = dots + (long long)row * JP;
  int tid = threadIdx.x;

  // per-thread sorted (desc) shortlist of its strided elements (<=5)
  float lv[5];
  int nl = 0;
#pragma unroll
  for (int s = 0; s < 5; ++s) {
    int j = tid + s * 256;
    if (j >= JP) break;
    bool valid = (j < JJ) && (j <= i + MEMK);
    float val = valid ? dr[j] : -FLT_MAX;
    int q = nl;
    while (q > 0 && lv[q - 1] < val) { lv[q] = lv[q - 1]; --q; }
    lv[q] = val;
    ++nl;
  }

  int head = 0;
  float kth = -FLT_MAX, rowmax = -FLT_MAX;
  for (int it = 0; it < TOPKK; ++it) {
    float h = (head < nl) ? lv[head] : -FLT_MAX;
    sv[tid] = h; si[tid] = tid;
    __syncthreads();
    for (int o = 128; o > 0; o >>= 1) {
      if (tid < o) {
        if (sv[tid + o] > sv[tid]) { sv[tid] = sv[tid + o]; si[tid] = si[tid + o]; }
      }
      __syncthreads();
    }
    float m = sv[0];
    int w = si[0];
    __syncthreads();
    if (it == 0) rowmax = m;
    kth = m;
    if (tid == w) ++head;
  }

  // masked softmax keeping scores >= kth
  float ev[5];
  float sum = 0.f;
#pragma unroll
  for (int s = 0; s < 5; ++s) {
    int j = tid + s * 256;
    if (j >= JP) { ev[s] = 0.f; continue; }
    bool valid = (j < JJ) && (j <= i + MEMK);
    float val = valid ? dr[j] : -FLT_MAX;
    float e = (valid && val >= kth) ? __expf(val - rowmax) : 0.f;
    ev[s] = e;
    sum += e;
  }
  sv[tid] = sum; __syncthreads();
  for (int o = 128; o > 0; o >>= 1) {
    if (tid < o) sv[tid] += sv[tid + o];
    __syncthreads();
  }
  float inv = 1.0f / sv[0];
  bf16_t* pr = probs + (long long)row * JP;
#pragma unroll
  for (int s = 0; s < 5; ++s) {
    int j = tid + s * 256;
    if (j < JP) pr[j] = (bf16_t)(ev[s] * inv);
  }
}

// ---------------- host orchestration ----------------
static inline int cdivi(long long a, long long b) { return (int)((a + b - 1) / b); }

extern "C" void kernel_launch(void* const* d_in, const int* in_sizes, int n_in,
                              void* d_out, int out_size, void* d_ws, size_t ws_size,
                              hipStream_t stream) {
  (void)in_sizes; (void)n_in; (void)out_size; (void)ws_size;
  const float* x      = (const float*)d_in[0];
  const float* ln1_g  = (const float*)d_in[1];
  const float* ln1_b  = (const float*)d_in[2];
  const float* wq     = (const float*)d_in[3];
  const float* wk     = (const float*)d_in[4];
  const float* wv     = (const float*)d_in[5];
  const float* pre_p  = (const float*)d_in[6];
  const float* post_p = (const float*)d_in[7];
  const float* mem_k  = (const float*)d_in[8];
  const float* mem_v  = (const float*)d_in[9];
  const float* wo     = (const float*)d_in[10];
  const float* bo     = (const float*)d_in[11];
  const float* ln2_g  = (const float*)d_in[12];
  const float* ln2_b  = (const float*)d_in[13];
  const float* w1     = (const float*)d_in[14];
  const float* b1     = (const float*)d_in[15];
  const float* w2     = (const float*)d_in[16];
  const float* b2     = (const float*)d_in[17];

  char* base = (char*)d_ws;
  size_t off = 0;
  auto alloc = [&](size_t bytes) -> void* {
    off = (off + 255) & ~(size_t)255;
    void* p = base + off;
    off += bytes;
    return p;
  };

  float*  xbuf    = (float*) alloc((size_t)TOK * DIMD * 4);
  bf16_t* hbuf    = (bf16_t*)alloc((size_t)TOK * DIMD * 2);
  bf16_t* wq_b    = (bf16_t*)alloc((size_t)DEPTHL * DIMD * INNER * 2);
  bf16_t* wk_b    = (bf16_t*)alloc((size_t)DEPTHL * DIMD * INNER * 2);
  bf16_t* wv_b    = (bf16_t*)alloc((size_t)DEPTHL * DIMD * INNER * 2);
  bf16_t* wo_b    = (bf16_t*)alloc((size_t)DEPTHL * INNER * DIMD * 2);
  bf16_t* w1_b    = (bf16_t*)alloc((size_t)DEPTHL * DIMD * DFF * 2);
  bf16_t* w2_b    = (bf16_t*)alloc((size_t)DEPTHL * DFF * DIMD * 2);
  bf16_t* qbuf    = (bf16_t*)alloc((size_t)TOK * INNER * 2);
  bf16_t* kbuf    = (bf16_t*)alloc((size_t)TOK * INNER * 2);
  bf16_t* vbuf    = (bf16_t*)alloc((size_t)TOK * INNER * 2);
  bf16_t* qh      = (bf16_t*)alloc((size_t)BB * HH * NN * DHD * 2);
  bf16_t* Kc      = (bf16_t*)alloc((size_t)BB * HH * JP * DHD * 2);
  bf16_t* Vc      = (bf16_t*)alloc((size_t)BB * HH * JP * DHD * 2);
  float*  dots    = (float*) alloc((size_t)BB * HH * NN * JP * 4);
  bf16_t* probs   = (bf16_t*)alloc((size_t)BB * HH * NN * JP * 2);
  bf16_t* attnout = (bf16_t*)alloc((size_t)TOK * INNER * 2);
  bf16_t* ffbuf   = (bf16_t*)alloc((size_t)TOK * DFF * 2);

  // cast all weights to bf16 once per launch (deterministic, no caching)
  {
    long long n = (long long)DEPTHL * DIMD * INNER;
    cast_bf16_kernel<<<cdivi(n, 256), 256, 0, stream>>>(wq, wq_b, n);
    cast_bf16_kernel<<<cdivi(n, 256), 256, 0, stream>>>(wk, wk_b, n);
    cast_bf16_kernel<<<cdivi(n, 256), 256, 0, stream>>>(wv, wv_b, n);
    cast_bf16_kernel<<<cdivi(n, 256), 256, 0, stream>>>(wo, wo_b, n);
    long long nf = (long long)DEPTHL * DIMD * DFF;
    cast_bf16_kernel<<<cdivi(nf, 256), 256, 0, stream>>>(w1, w1_b, nf);
    cast_bf16_kernel<<<cdivi(nf, 256), 256, 0, stream>>>(w2, w2_b, nf);
  }

  hipMemcpyAsync(xbuf, x, (size_t)TOK * DIMD * 4, hipMemcpyDeviceToDevice, stream);

  dim3 gProj(cdivi(TOK, BM), cdivi(INNER, BN), 1);     // 32 x 8
  dim3 gFF1(cdivi(TOK, BM), cdivi(DFF, BN), 1);        // 32 x 32
  dim3 gScore(cdivi(NN, BM), cdivi(JP, BN), BB * HH);  // 8 x 9 x 64
  dim3 gAV(cdivi(NN, BM), cdivi(DHD, BN), BB * HH);    // 8 x 1 x 64

  for (int l = 0; l < DEPTHL; ++l) {
    // ---- attention block ----
    ln_kernel<<<TOK, 256, 0, stream>>>(xbuf, ln1_g + l * DIMD, ln1_b + l * DIMD, hbuf);

    gemm_wmma<<<gProj, 256, 0, stream>>>(hbuf, 0, DIMD,
        wq_b + (size_t)l * DIMD * INNER, 0, INNER, 0,
        nullptr, qbuf, 1, 0, 0, INNER, nullptr, nullptr,
        TOK, INNER, DIMD, 1.0f, 0);
    gemm_wmma<<<gProj, 256, 0, stream>>>(hbuf, 0, DIMD,
        wk_b + (size_t)l * DIMD * INNER, 0, INNER, 0,
        nullptr, kbuf, 1, 0, 0, INNER, nullptr, nullptr,
        TOK, INNER, DIMD, 1.0f, 0);
    gemm_wmma<<<gProj, 256, 0, stream>>>(hbuf, 0, DIMD,
        wv_b + (size_t)l * DIMD * INNER, 0, INNER, 0,
        nullptr, vbuf, 1, 0, 0, INNER, nullptr, nullptr,
        TOK, INNER, DIMD, 1.0f, 0);

    split_q_kernel<<<cdivi((long long)BB * HH * NN * DHD, 256), 256, 0, stream>>>(qbuf, qh);
    build_kv_kernel<<<cdivi((long long)BB * HH * JP * DHD, 256), 256, 0, stream>>>(
        kbuf, vbuf, mem_k, mem_v, l, Kc, Vc);

    // scores: dots[b,h,i,j] = SCALE * q . k   (B^T load path, K=DH=64)
    gemm_wmma<<<gScore, 256, 0, stream>>>(qh, (long long)NN * DHD, DHD,
        Kc, (long long)JP * DHD, DHD, 1,
        dots, nullptr, 1, (long long)NN * JP, 0, JP, nullptr, nullptr,
        NN, JP, DHD, SCALEF, 0);

    talking_kernel<float><<<cdivi((long long)BB * NN * JP, 256), 256, 0, stream>>>(
        dots, pre_p + l * HH * HH);

    topk_softmax_kernel<<<BB * HH * NN, 256, 0, stream>>>(dots, probs);

    talking_kernel<bf16_t><<<cdivi((long long)BB * NN * JP, 256), 256, 0, stream>>>(
        probs, post_p + l * HH * HH);

    // attn @ V -> attnout laid out as [b, i, h*64+d] via split z-strides
    gemm_wmma<<<gAV, 256, 0, stream>>>(probs, (long long)NN * JP, JP,
        Vc, (long long)JP * DHD, DHD, 0,
        nullptr, attnout, HH, (long long)NN * INNER, DHD, INNER, nullptr, nullptr,
        NN, DHD, JP, 1.0f, 0);

    // output projection + bias + residual -> xbuf (f32)
    gemm_wmma<<<gProj, 256, 0, stream>>>(attnout, 0, INNER,
        wo_b + (size_t)l * INNER * DIMD, 0, DIMD, 0,
        xbuf, nullptr, 1, 0, 0, DIMD, bo + l * DIMD, xbuf,
        TOK, DIMD, INNER, 1.0f, 0);

    // ---- feedforward block ----
    ln_kernel<<<TOK, 256, 0, stream>>>(xbuf, ln2_g + l * DIMD, ln2_b + l * DIMD, hbuf);

    gemm_wmma<<<gFF1, 256, 0, stream>>>(hbuf, 0, DIMD,
        w1_b + (size_t)l * DIMD * DFF, 0, DFF, 0,
        nullptr, ffbuf, 1, 0, 0, DFF, b1 + l * DFF, nullptr,
        TOK, DFF, DIMD, 1.0f, 1);

    gemm_wmma<<<gProj, 256, 0, stream>>>(ffbuf, 0, DFF,
        w2_b + (size_t)l * DFF * DIMD, 0, DIMD, 0,
        xbuf, nullptr, 1, 0, 0, DIMD, b2 + l * DIMD, xbuf,
        TOK, DIMD, DFF, 1.0f, 0);
  }

  hipMemcpyAsync(d_out, xbuf, (size_t)TOK * DIMD * 4, hipMemcpyDeviceToDevice, stream);
}
